// VNETDetector_62947040690718
// MI455X (gfx1250) — compile-verified, matrix operand order checked by reading, not run on hardware
//
#include <hip/hip_runtime.h>
#include <hip/hip_bf16.h>
#include <math.h>

typedef _Float16 v16h __attribute__((ext_vector_type(16)));
typedef float    v8f  __attribute__((ext_vector_type(8)));
typedef float    v16f __attribute__((ext_vector_type(16)));

#define L_TOTAL 262144
#define NSTATES 16
#define H1S 100
#define H2S 50
#define CHUNK 128
#define NCHUNKS (L_TOTAL / CHUNK)   /* 2048 */

__device__ __forceinline__ int kmap(int e) { return (e < 8) ? e : (e + 8); }

// ---------------------------------------------------------------------------
// Kernel 0: pack weights into WMMA fragment layout (runs once per launch).
//   b2f: 16 frags (nt,kc) x 32 lanes x 16 f16      (w2^T tiles, padded)
//   b3f:  2 frags (kh)    x 32 lanes x 16 f16      (w3^T tiles, padded)
//   w1p/b1p: 4 frags (kc) x 32 lanes x 16 f32      (layer-1 taps, padded;
//            pad bias = -60 so sigmoid -> 0 exactly in f16)
//   b2p: b2 zero-padded to 64
// ---------------------------------------------------------------------------
__global__ void vnet_pack_weights(const float* __restrict__ w1,
                                  const float* __restrict__ b1,
                                  const float* __restrict__ w2,
                                  const float* __restrict__ b2,
                                  const float* __restrict__ w3,
                                  _Float16* __restrict__ b2f,
                                  _Float16* __restrict__ b3f,
                                  float* __restrict__ w1p,
                                  float* __restrict__ b1p,
                                  float* __restrict__ b2p) {
    const int tid = threadIdx.x;
    for (int i = tid; i < 16 * 512; i += 256) {           // w2 fragments
        const int f = i >> 9, l = (i >> 4) & 31, e = i & 15;
        const int nt = f >> 2, kc = f & 3;
        const int hi = l >> 4, col = nt * 16 + (l & 15);
        const int k = kc * 32 + hi * 8 + kmap(e);
        const float v = (k < H1S && col < H2S) ? w2[col * H1S + k] : 0.0f;
        b2f[i] = (_Float16)v;
    }
    for (int i = tid; i < 2 * 512; i += 256) {            // w3 fragments
        const int f = i >> 9, l = (i >> 4) & 31, e = i & 15;
        const int hi = l >> 4, n = l & 15;
        const int k = f * 32 + hi * 8 + kmap(e);
        const float v = (k < H2S) ? w3[n * H2S + k] : 0.0f;
        b3f[i] = (_Float16)v;
    }
    for (int i = tid; i < 4 * 512; i += 256) {            // layer-1 taps
        const int kc = i >> 9, l = (i >> 4) & 31, e = i & 15;
        const int hi = l >> 4;
        const int k = kc * 32 + hi * 8 + kmap(e);
        w1p[i] = (k < H1S) ? w1[k] : 0.0f;
        b1p[i] = (k < H1S) ? b1[k] : -60.0f;
    }
    for (int i = tid; i < 64; i += 256)
        b2p[i] = (i < H2S) ? b2[i] : 0.0f;
}

// ---------------------------------------------------------------------------
// Kernel 1: MLP -> llr table (L x 16), WMMA f16->f32.
// One wave per 16-row tile of L; 8 waves / block. All weight fragments come
// pre-packed, so every VMEM access is a contiguous b128.
// ---------------------------------------------------------------------------
__global__ __launch_bounds__(256)
void vnet_mlp_wmma(const float* __restrict__ y,
                   const _Float16* __restrict__ b2f,
                   const _Float16* __restrict__ b3f,
                   const float* __restrict__ w1p,
                   const float* __restrict__ b1p,
                   const float* __restrict__ b2p,
                   const float* __restrict__ b3,
                   float* __restrict__ llr) {
    __shared__ _Float16 h2s[8][16][64];
    __shared__ float    llds[8][256];
    const int lane = threadIdx.x & 31;
    const int wid  = threadIdx.x >> 5;
    const int tile = blockIdx.x * 8 + wid;
    const int m    = lane & 15;
    const int hi   = (lane >> 4) & 1;

    const float yv = y[tile * 16 + m];
    const v16h* b2fv = (const v16h*)b2f;
    const v16h* b3fv = (const v16h*)b3f;
    const v16f* w1pv = (const v16f*)w1p;
    const v16f* b1pv = (const v16f*)b1p;

    // ---- Layer 1 straight into WMMA A fragments (sigmoid via v_rcp_f32).
    v16h a2[4];
    #pragma unroll
    for (int kc = 0; kc < 4; ++kc) {
        const v16f wv = w1pv[kc * 32 + lane];
        const v16f bv = b1pv[kc * 32 + lane];
        #pragma unroll
        for (int e = 0; e < 16; ++e) {
            const float z = fmaf(yv, wv[e], bv[e]);
            const float s = __builtin_amdgcn_rcpf(1.0f + __expf(-z));
            a2[kc][e] = (_Float16)s;
        }
    }

    // ---- Layer 2: h2 = relu(h1 @ w2^T + b2); 4 N-tiles x 4 K-steps.
    #pragma unroll
    for (int nt = 0; nt < 4; ++nt) {
        const int col = nt * 16 + m;
        const float bias = b2p[col];
        v8f c;
        #pragma unroll
        for (int j = 0; j < 8; ++j) c[j] = bias;
        #pragma unroll
        for (int kc = 0; kc < 4; ++kc) {
            const v16h b = b2fv[(nt * 4 + kc) * 32 + lane];
            c = __builtin_amdgcn_wmma_f32_16x16x32_f16(false, a2[kc], false, b,
                                                       (short)0, c, false, false);
        }
        #pragma unroll
        for (int j = 0; j < 8; ++j) {
            float v = c[j];
            v = v > 0.0f ? v : 0.0f;
            h2s[wid][j + hi * 8][col] = (_Float16)v;
        }
    }
    __syncthreads();

    // ---- Layer 3: priors = h2 @ w3^T + b3; 2 K-steps, single N tile.
    v8f c3;
    const float bias3 = b3[m];
    #pragma unroll
    for (int j = 0; j < 8; ++j) c3[j] = bias3;
    #pragma unroll
    for (int kh = 0; kh < 2; ++kh) {
        const int kb = kh * 32 + hi * 8;
        v16h a;
        #pragma unroll
        for (int e = 0; e < 16; ++e) a[e] = h2s[wid][m][kb + kmap(e)];
        const v16h b = b3fv[kh * 32 + lane];
        c3 = __builtin_amdgcn_wmma_f32_16x16x32_f16(false, a, false, b,
                                                    (short)0, c3, false, false);
    }

    // ---- llr = -priors; repack through LDS for coalesced b128 stores.
    #pragma unroll
    for (int j = 0; j < 8; ++j) llds[wid][(j + hi * 8) * 16 + m] = -c3[j];
    __syncthreads();
    const float4* src = (const float4*)&llds[wid][lane * 8];
    float4* dst = (float4*)(llr + (size_t)tile * 256 + lane * 8);
    dst[0] = src[0];
    dst[1] = src[1];
}

// ---------------------------------------------------------------------------
// Kernel 2: per-chunk 16x16 min-plus transition matrices.
// 16 lanes = 16 matrix columns; two chunks per wave32.
// ---------------------------------------------------------------------------
__global__ __launch_bounds__(256)
void vnet_chunk_mats(const float* __restrict__ llr, float* __restrict__ cm) {
    const int lane  = threadIdx.x & 31;
    const int wid   = threadIdx.x >> 5;
    const int grp   = lane >> 4;
    const int p     = lane & 15;
    const int chunk = (blockIdx.x * 8 + wid) * 2 + grp;

    float r[16];
    #pragma unroll
    for (int i = 0; i < 16; ++i) r[i] = (i == p) ? 0.0f : 1e30f;

    const float* base = llr + (size_t)chunk * CHUNK * NSTATES;
    for (int t = 0; t < CHUNK; ++t) {
        const float lv = base[t * NSTATES + p];
        float ls[16];
        #pragma unroll
        for (int q = 0; q < 16; ++q) ls[q] = __shfl(lv, q, 16);
        float nr[16];
        #pragma unroll
        for (int nx = 0; nx < 16; ++nx) {
            const int q0 = nx >> 1, q1 = q0 + 8;
            const float m0 = r[q0] + ls[q0];
            const float m1 = r[q1] + ls[q1];
            nr[nx] = m0 < m1 ? m0 : m1;
        }
        #pragma unroll
        for (int i = 0; i < 16; ++i) r[i] = nr[i];
    }
    float* out = cm + (size_t)chunk * 256;
    #pragma unroll
    for (int nx = 0; nx < 16; ++nx) out[nx * 16 + p] = r[nx];
}

// ---------------------------------------------------------------------------
// Kernel 3: serial min-plus prefix over chunk matrices (one wave).
// ---------------------------------------------------------------------------
__global__ void vnet_scan_chunks(const float* __restrict__ cm,
                                 float* __restrict__ cv) {
    const int lane = threadIdx.x & 31;
    const int n = lane & 15;
    float v = 0.0f;
    for (int c = 0; c < NCHUNKS; ++c) {
        if (lane < 16) cv[c * NSTATES + n] = v;
        const float* M = cm + (size_t)c * 256 + n * 16;
        float nv = 1e30f;
        #pragma unroll
        for (int q = 0; q < 16; ++q) {
            const float s = M[q] + __shfl(v, q, 16);
            nv = s < nv ? s : nv;
        }
        v = nv;
    }
}

// ---------------------------------------------------------------------------
// Kernel 4: replay each chunk with its entry vector; emit bits.
// ---------------------------------------------------------------------------
__global__ __launch_bounds__(256)
void vnet_replay(const float* __restrict__ llr, const float* __restrict__ cv,
                 float* __restrict__ out) {
    const int chunk = blockIdx.x * blockDim.x + threadIdx.x;
    if (chunk >= NCHUNKS) return;
    float v[16];
    #pragma unroll
    for (int i = 0; i < 16; ++i) v[i] = cv[chunk * NSTATES + i];
    const float4* lb = (const float4*)(llr + (size_t)chunk * CHUNK * NSTATES);
    float* ob = out + (size_t)chunk * CHUNK;
    for (int t = 0; t < CHUNK; ++t) {
        float mv = v[0]; int mi = 0;
        #pragma unroll
        for (int i = 1; i < 16; ++i) { if (v[i] < mv) { mv = v[i]; mi = i; } }
        ob[t] = (float)(mi & 1);
        float l[16];
        #pragma unroll
        for (int q = 0; q < 4; ++q) {
            const float4 f = lb[t * 4 + q];
            l[q*4+0]=f.x; l[q*4+1]=f.y; l[q*4+2]=f.z; l[q*4+3]=f.w;
        }
        float nv[16];
        #pragma unroll
        for (int nx = 0; nx < 16; ++nx) {
            const int q0 = nx >> 1, q1 = q0 + 8;
            const float m0 = v[q0] + l[q0];
            const float m1 = v[q1] + l[q1];
            nv[nx] = m0 < m1 ? m0 : m1;
        }
        #pragma unroll
        for (int i = 0; i < 16; ++i) v[i] = nv[i];
    }
}

extern "C" void kernel_launch(void* const* d_in, const int* in_sizes, int n_in,
                              void* d_out, int out_size, void* d_ws, size_t ws_size,
                              hipStream_t stream) {
    const float* y  = (const float*)d_in[0];
    const float* w1 = (const float*)d_in[1];
    const float* b1 = (const float*)d_in[2];
    const float* w2 = (const float*)d_in[3];
    const float* b2 = (const float*)d_in[4];
    const float* w3 = (const float*)d_in[5];
    const float* b3 = (const float*)d_in[6];

    char* ws = (char*)d_ws;
    size_t off = 0;
    float* llr = (float*)(ws + off); off += (size_t)L_TOTAL * NSTATES * 4;   // 16 MB
    float* cm  = (float*)(ws + off); off += (size_t)NCHUNKS * 256 * 4;       //  2 MB
    float* cv  = (float*)(ws + off); off += (size_t)NCHUNKS * NSTATES * 4;   // 128 KB
    _Float16* b2f = (_Float16*)(ws + off); off += 16 * 512 * sizeof(_Float16);
    _Float16* b3f = (_Float16*)(ws + off); off +=  2 * 512 * sizeof(_Float16);
    float* w1p = (float*)(ws + off); off += 4 * 512 * sizeof(float);
    float* b1p = (float*)(ws + off); off += 4 * 512 * sizeof(float);
    float* b2p = (float*)(ws + off); off += 64 * sizeof(float);

    vnet_pack_weights<<<1, 256, 0, stream>>>(w1, b1, w2, b2, w3,
                                             b2f, b3f, w1p, b1p, b2p);
    vnet_mlp_wmma<<<L_TOTAL / 16 / 8, 256, 0, stream>>>(y, b2f, b3f, w1p, b1p,
                                                        b2p, b3, llr);
    vnet_chunk_mats<<<NCHUNKS / 16, 256, 0, stream>>>(llr, cm);
    vnet_scan_chunks<<<1, 32, 0, stream>>>(cm, cv);
    vnet_replay<<<(NCHUNKS + 255) / 256, 256, 0, stream>>>(llr, cv, (float*)d_out);
}
